// LSTM_37160057045197
// MI455X (gfx1250) — compile-verified, matrix operand order checked by reading, not run on hardware
//
#include <hip/hip_runtime.h>
#include <hip/hip_bf16.h>

#define LSTM_UNITS 64
#define LSTM_T     100
#define LSTM_F     3
#define GATES      256   // 4*UNITS

typedef __attribute__((ext_vector_type(16))) _Float16 v16h;
typedef __attribute__((ext_vector_type(8)))  _Float16 v8h;
typedef __attribute__((ext_vector_type(4)))  _Float16 v4h;
typedef __attribute__((ext_vector_type(8)))  float    v8f;

__device__ __forceinline__ float fsig(float x)  { return 1.0f / (1.0f + __expf(-x)); }
__device__ __forceinline__ float ftanhf(float x){ return 1.0f - 2.0f / (1.0f + __expf(2.0f * x)); }

__global__ __launch_bounds__(128)
void lstm_wmma_kernel(const float* __restrict__ x,   // [B, T, F]
                      const float* __restrict__ W,   // [F, 256]
                      const float* __restrict__ U,   // [64, 256]
                      const float* __restrict__ b,   // [256]
                      const float* __restrict__ W1,  // [64, 64]
                      const float* __restrict__ b1,  // [64]
                      const float* __restrict__ W2,  // [64, 5]
                      const float* __restrict__ b2,  // [5]
                      float* __restrict__ out)       // [B, 5]
{
    // LDS: x tile as f16 [m][t][4] (x0,x1,x2,1.0); h tile [m][n] f16; head scratch.
    __shared__ __align__(16) _Float16 x_lds[16 * LSTM_T * 4];
    __shared__ __align__(16) _Float16 h_lds[16 * LSTM_UNITS];
    __shared__ __align__(16) float    h1_lds[16 * LSTM_UNITS];
    __shared__ __align__(16) float    logit_lds[16 * 5];

    const int tid  = threadIdx.x;
    const int wave = tid >> 5;     // 0..3 : owns unit slice [16*wave, 16*wave+16)
    const int lane = tid & 31;
    const int m    = lane & 15;    // batch column within tile
    const int hi   = lane >> 4;    // lane half
    const int tileBase = blockIdx.x * 16;

    // ---- stage x tile into LDS as f16, 4th slot = 1.0 (bias column of B) ----
    for (int idx = tid; idx < 16 * LSTM_T * LSTM_F; idx += 128) {
        int mm = idx / (LSTM_T * LSTM_F);
        int rr = idx - mm * (LSTM_T * LSTM_F);
        int tt = rr / LSTM_F;
        int ff = rr - tt * LSTM_F;
        float v = x[(size_t)(tileBase + mm) * (LSTM_T * LSTM_F) + rr];
        x_lds[(mm * LSTM_T + tt) * 4 + ff] = (_Float16)v;
    }
    for (int idx = tid; idx < 16 * LSTM_T; idx += 128)
        x_lds[idx * 4 + 3] = (_Float16)1.0f;
    for (int idx = tid; idx < 16 * LSTM_UNITS; idx += 128)
        h_lds[idx] = (_Float16)0.0f;

    // ---- persistent A fragments: A_aug[g,k] = U[k,g] | W[k-64,g] | b[g] | 0 ----
    // A 16-bit layout: lane(0-15)=row M, halves j -> k = (j&7) + 8*hi + 16*(j>>3) + 32*kf
    v16h aF[4][3];
#pragma unroll
    for (int c = 0; c < 4; ++c) {
        const int g = 16 * (wave + 4 * c) + m;   // gate row (c=0:i,1:f,2:g,3:o)
#pragma unroll
        for (int kf = 0; kf < 3; ++kf) {
            v16h a;
#pragma unroll
            for (int j = 0; j < 16; ++j) {
                int k = (j & 7) + 8 * hi + 16 * (j >> 3) + 32 * kf;
                float v;
                if (k < LSTM_UNITS)          v = U[k * GATES + g];
                else if (k < LSTM_UNITS + 3) v = W[(k - LSTM_UNITS) * GATES + g];
                else if (k == LSTM_UNITS + 3)v = b[g];
                else                         v = 0.0f;
                a[j] = (_Float16)v;
            }
            aF[c][kf] = a;
        }
    }

    v8f cst   = {};   // cell state, wave's unit slice, D layout
    v8f hprev = {};   // masked h,   same layout
    __syncthreads();

    const _Float16* hrow   = &h_lds[m * LSTM_UNITS + 8 * hi];
    _Float16*       hstore = &h_lds[m * LSTM_UNITS + 16 * wave + 8 * hi];

    for (int t = 0; t < LSTM_T; ++t) {
        // B fragments for h (k = unit index): lane-local b128 loads
        v8h p0 = *(const v8h*)(hrow + 0);
        v8h p1 = *(const v8h*)(hrow + 16);
        v8h p2 = *(const v8h*)(hrow + 32);
        v8h p3 = *(const v8h*)(hrow + 48);
        v16h b0f = __builtin_shufflevector(p0, p1, 0,1,2,3,4,5,6,7,8,9,10,11,12,13,14,15);
        v16h b1f = __builtin_shufflevector(p2, p3, 0,1,2,3,4,5,6,7,8,9,10,11,12,13,14,15);

        // B fragment for [x0,x1,x2,1,0,...] (k=64..95); only lanes hi==0, j<4 nonzero
        v4h x4 = *(const v4h*)&x_lds[(m * LSTM_T + t) * 4];
        bool maskb = ((float)x4[0] != 0.0f) || ((float)x4[1] != 0.0f) || ((float)x4[2] != 0.0f);
        const _Float16 z0 = (_Float16)0.0f;
        v16h b2f = { hi ? z0 : x4[0], hi ? z0 : x4[1], hi ? z0 : x4[2], hi ? z0 : x4[3],
                     z0, z0, z0, z0, z0, z0, z0, z0, z0, z0, z0, z0 };

        // z^T tiles for i,f,g,o on this wave's 16-unit slice
        v8f acc[4];
#pragma unroll
        for (int c = 0; c < 4; ++c) {
            v8f a = {};
            a = __builtin_amdgcn_wmma_f32_16x16x32_f16(false, aF[c][2], false, b2f, (short)0, a, false, false);
            a = __builtin_amdgcn_wmma_f32_16x16x32_f16(false, aF[c][0], false, b0f, (short)0, a, false, false);
            a = __builtin_amdgcn_wmma_f32_16x16x32_f16(false, aF[c][1], false, b1f, (short)0, a, false, false);
            acc[c] = a;
        }

        // gate math: element (unit n = 16*wave + r + 8*hi, batch col m)
#pragma unroll
        for (int r = 0; r < 8; ++r) {
            float iv = fsig(acc[0][r]);
            float fv = fsig(acc[1][r]);
            float gv = ftanhf(acc[2][r]);
            float ov = fsig(acc[3][r]);
            float cn = fv * cst[r] + iv * gv;
            float hn = ov * ftanhf(cn);
            cst[r]   = maskb ? cn : cst[r];
            hprev[r] = maskb ? hn : hprev[r];
        }

        __syncthreads();                    // everyone done reading old h
        v8h hh;
#pragma unroll
        for (int r = 0; r < 8; ++r) hh[r] = (_Float16)hprev[r];
        *(v8h*)hstore = hh;                 // publish new h (8 consecutive units)
        __syncthreads();                    // new h visible for next step
    }

    // ---- head: h1 = relu(h @ W1 + b1); logits = h1 @ W2 + b2; softmax ----
    {
        const int mm = tid & 15;
        const int qg = tid >> 4;            // 0..7
#pragma unroll
        for (int qq = 0; qq < 8; ++qq) {
            int q = qg * 8 + qq;
            float acc = b1[q];
            for (int n = 0; n < LSTM_UNITS; ++n)
                acc += (float)h_lds[mm * LSTM_UNITS + n] * W1[n * LSTM_UNITS + q];
            h1_lds[mm * LSTM_UNITS + q] = fmaxf(acc, 0.0f);
        }
    }
    __syncthreads();
    if (tid < 80) {
        const int mm  = tid & 15;
        const int cls = tid >> 4;           // 0..4
        float acc = b2[cls];
        for (int n = 0; n < LSTM_UNITS; ++n)
            acc += h1_lds[mm * LSTM_UNITS + n] * W2[n * 5 + cls];
        logit_lds[mm * 5 + cls] = acc;
    }
    __syncthreads();
    if (tid < 16) {
        float l[5], mx = -3.0e38f;
#pragma unroll
        for (int c = 0; c < 5; ++c) { l[c] = logit_lds[tid * 5 + c]; mx = fmaxf(mx, l[c]); }
        float s = 0.0f;
#pragma unroll
        for (int c = 0; c < 5; ++c) { l[c] = __expf(l[c] - mx); s += l[c]; }
        float inv = 1.0f / s;
#pragma unroll
        for (int c = 0; c < 5; ++c)
            out[(size_t)(tileBase + tid) * 5 + c] = l[c] * inv;
    }
}

extern "C" void kernel_launch(void* const* d_in, const int* in_sizes, int n_in,
                              void* d_out, int out_size, void* d_ws, size_t ws_size,
                              hipStream_t stream) {
    const float* x  = (const float*)d_in[0];
    const float* W  = (const float*)d_in[1];
    const float* U  = (const float*)d_in[2];
    const float* b  = (const float*)d_in[3];
    const float* W1 = (const float*)d_in[4];
    const float* b1 = (const float*)d_in[5];
    const float* W2 = (const float*)d_in[6];
    const float* b2 = (const float*)d_in[7];
    float* out = (float*)d_out;

    const int B = in_sizes[0] / (LSTM_T * LSTM_F);   // 16384
    const int nTiles = B / 16;                        // 1024
    lstm_wmma_kernel<<<dim3(nTiles), dim3(128), 0, stream>>>(x, W, U, b, W1, b1, W2, b2, out);
}
